// RoutedExpertCore_41102837023409
// MI455X (gfx1250) — matrix operand order repaired and checked by reading, not compiled, not run on hardware
//
#include <hip/hip_runtime.h>
#include <math.h>

#define NB   8
#define SEQ  4096
#define NTOK (NB*SEQ)      // 32768 tokens
#define DD   128
#define HH   128
#define FF   256
#define EE   32
#define MAX_TILES 4128     // 2*NTOK/16 + EE

typedef __attribute__((ext_vector_type(16))) __bf16 v16bf;
typedef __attribute__((ext_vector_type(8)))  float  v8f;
typedef __attribute__((ext_vector_type(4)))  int    v4i;

static __device__ __forceinline__ __bf16 f2bf(float f) {
  unsigned u = __builtin_bit_cast(unsigned, f);
  u += 0x7FFFu + ((u >> 16) & 1u);                 // round-to-nearest-even
  unsigned short s = (unsigned short)(u >> 16);
  return __builtin_bit_cast(__bf16, s);
}

// A fragment: 16(M) x 32(K) bf16, row-major source, stride lda (elements).
// lane<16 : row=lane,    elems 0..7 = K[0..7],  8..15 = K[16..23]
// lane>=16: row=lane-16, elems 0..7 = K[8..15], 8..15 = K[24..31]
static __device__ __forceinline__ v16bf load_frag_a(const __bf16* A, int lda) {
  int lane = threadIdx.x & 31;
  int row  = lane & 15;
  int hb   = (lane >> 4) << 3;           // 0 or 8
  const __bf16* p0 = A + row * lda + hb;
  const __bf16* p1 = p0 + 16;
  v16bf a;
#pragma unroll
  for (int j = 0; j < 8; ++j) { a[j] = p0[j]; a[8 + j] = p1[j]; }
  return a;
}

// B fragment: 32(K) x 16(N), supplied as BT (N x K row-major), stride ldb.
// lane<16 : col=lane,    elems 0..7 = K[0..7],  8..15 = K[16..23]
// lane>=16: col=lane-16, elems 0..7 = K[8..15], 8..15 = K[24..31]
static __device__ __forceinline__ v16bf load_frag_bt(const __bf16* BT, int ldb) {
  int lane = threadIdx.x & 31;
  int n  = lane & 15;
  int kb = (lane >> 4) << 3;             // 0 or 8
  const __bf16* p = BT + n * ldb + kb;
  v16bf b;
#pragma unroll
  for (int j = 0; j < 8; ++j) { b[j] = p[j]; b[8 + j] = p[16 + j]; }
  return b;
}

static __device__ __forceinline__ v8f wmma_bf16(v16bf a, v16bf b, v8f c) {
  return __builtin_amdgcn_wmma_f32_16x16x32_bf16(false, a, false, b, (short)0, c, false, false);
}

// ---------------- prep: bf16 conversions / transposes / init ----------------
__global__ void k_conv_small(const float* __restrict__ Wp, const float* __restrict__ Wq,
                             const float* __restrict__ keys,
                             __bf16* WpB, __bf16* WqB, __bf16* keysB, int* counts) {
  int i = blockIdx.x * blockDim.x + threadIdx.x;
  if (i < HH * DD) WpB[i]   = f2bf(Wp[i]);
  if (i < HH * HH) WqB[i]   = f2bf(Wq[i]);
  if (i < EE * HH) keysB[i] = f2bf(keys[i]);
  if (i < EE)      counts[i] = 0;
}

__global__ void k_t_w1(const float* __restrict__ W1, __bf16* __restrict__ W1T) {
  int i = blockIdx.x * blockDim.x + threadIdx.x;          // over E*F*H, layout [e][f][h]
  if (i >= EE * FF * HH) return;
  int e = i / (FF * HH), rem = i % (FF * HH);
  int f = rem / HH, h = rem % HH;
  W1T[i] = f2bf(W1[(size_t)e * HH * FF + (size_t)h * FF + f]);
}

__global__ void k_t_w2(const float* __restrict__ W2, __bf16* __restrict__ W2T) {
  int i = blockIdx.x * blockDim.x + threadIdx.x;          // over E*H*F, layout [e][h][f]
  if (i >= EE * HH * FF) return;
  int e = i / (HH * FF), rem = i % (HH * FF);
  int h = rem / FF, f = rem % FF;
  W2T[i] = f2bf(W2[(size_t)e * FF * HH + (size_t)f * HH + h]);
}

// ---------------- projection: tokens = tokens_in @ Wp^T + bp ----------------
__global__ void __launch_bounds__(128)
k_proj(const float* __restrict__ tokens_in, const __bf16* __restrict__ WpT,
       const float* __restrict__ bp, float* __restrict__ tok_f, __bf16* __restrict__ tok_bf) {
  __shared__ __bf16 As[64 * DD];
  int wg = blockIdx.x, tid = threadIdx.x;                 // 512 wgs x 64 tokens
  const float* src = tokens_in + (size_t)wg * 64 * DD;
  for (int i = tid; i < 64 * DD; i += 128) As[i] = f2bf(src[i]);
  __syncthreads();
  int wave = tid >> 5, lane = tid & 31;
  v8f acc[8];
#pragma unroll
  for (int nt = 0; nt < 8; ++nt)
#pragma unroll
    for (int r = 0; r < 8; ++r) acc[nt][r] = 0.f;
#pragma unroll
  for (int ks = 0; ks < 4; ++ks) {
    v16bf a = load_frag_a(As + (wave * 16) * DD + ks * 32, DD);
#pragma unroll
    for (int nt = 0; nt < 8; ++nt) {
      v16bf b = load_frag_bt(WpT + nt * 16 * DD + ks * 32, DD);
      acc[nt] = wmma_bf16(a, b, acc[nt]);
    }
  }
  int col0 = lane & 15, rbase = (lane >> 4) << 3;
#pragma unroll
  for (int nt = 0; nt < 8; ++nt)
#pragma unroll
    for (int r = 0; r < 8; ++r) {
      int row = rbase + r, col = nt * 16 + col0;
      float v = acc[nt][r] + bp[col];
      size_t g = (size_t)(wg * 64 + wave * 16 + row) * HH + col;
      tok_f[g] = v;
      tok_bf[g] = f2bf(v);
    }
}

// ------------- routing: query GEMM, score GEMM, softmax top-2 --------------
__global__ void __launch_bounds__(128)
k_route(const __bf16* __restrict__ tok_bf, const __bf16* __restrict__ WqT,
        const __bf16* __restrict__ keysT, int* __restrict__ route_idx,
        float* __restrict__ route_w, int* __restrict__ counts) {
  __shared__ __bf16 qs[64 * HH];
  __shared__ float  ss[64 * EE];
  int wg = blockIdx.x, tid = threadIdx.x, wave = tid >> 5, lane = tid & 31;
  size_t base = (size_t)wg * 64;
  int col0 = lane & 15, rbase = (lane >> 4) << 3;
  // query = tokens @ Wq^T
  v8f acc[8];
#pragma unroll
  for (int nt = 0; nt < 8; ++nt)
#pragma unroll
    for (int r = 0; r < 8; ++r) acc[nt][r] = 0.f;
#pragma unroll
  for (int ks = 0; ks < 4; ++ks) {
    v16bf a = load_frag_a(tok_bf + (base + wave * 16) * HH + ks * 32, HH);
#pragma unroll
    for (int nt = 0; nt < 8; ++nt) {
      v16bf b = load_frag_bt(WqT + nt * 16 * HH + ks * 32, HH);
      acc[nt] = wmma_bf16(a, b, acc[nt]);
    }
  }
#pragma unroll
  for (int nt = 0; nt < 8; ++nt)
#pragma unroll
    for (int r = 0; r < 8; ++r)
      qs[(wave * 16 + rbase + r) * HH + nt * 16 + col0] = f2bf(acc[nt][r]);
  __syncthreads();
  // scores = query @ keys^T / sqrt(H)
  v8f sacc[2];
#pragma unroll
  for (int nt = 0; nt < 2; ++nt)
#pragma unroll
    for (int r = 0; r < 8; ++r) sacc[nt][r] = 0.f;
#pragma unroll
  for (int ks = 0; ks < 4; ++ks) {
    v16bf a = load_frag_a(qs + (wave * 16) * HH + ks * 32, HH);
#pragma unroll
    for (int nt = 0; nt < 2; ++nt) {
      v16bf b = load_frag_bt(keysT + nt * 16 * HH + ks * 32, HH);
      sacc[nt] = wmma_bf16(a, b, sacc[nt]);
    }
  }
  const float scale = 0.08838834764831845f;   // 1/sqrt(128)
#pragma unroll
  for (int nt = 0; nt < 2; ++nt)
#pragma unroll
    for (int r = 0; r < 8; ++r)
      ss[(wave * 16 + rbase + r) * EE + nt * 16 + col0] = sacc[nt][r] * scale;
  __syncthreads();
  // top-2 + renormalized softmax weights (lowest index wins ties, as jax top_k)
  if (tid < 64) {
    const float* s = ss + tid * EE;
    float m1 = -3.4e38f; int i1 = 0;
    for (int e2 = 0; e2 < EE; ++e2) if (s[e2] > m1) { m1 = s[e2]; i1 = e2; }
    float m2 = -3.4e38f; int i2 = (i1 == 0) ? 1 : 0;
    for (int e2 = 0; e2 < EE; ++e2) if (e2 != i1 && s[e2] > m2) { m2 = s[e2]; i2 = e2; }
    float p1 = __expf(m2 - m1);                 // <= 1
    float inv = 1.f / (1.f + p1);
    int g = (int)base + tid;
    route_idx[2 * g]     = i1; route_idx[2 * g + 1] = i2;
    route_w[2 * g]       = inv; route_w[2 * g + 1]  = p1 * inv;
    atomicAdd(&counts[i1], 1);
    atomicAdd(&counts[i2], 1);
  }
}

// ------------- expert grouping: scan + tiles, scatter ----------------------
__global__ void k_scan(const int* __restrict__ counts, int* offsets, int* cursor,
                       int* tile_e, int* tile_s, int* n_tiles) {
  if (blockIdx.x == 0 && threadIdx.x == 0) {
    int acc = 0;
    for (int e = 0; e < EE; ++e) { offsets[e] = acc; cursor[e] = acc; acc += counts[e]; }
    offsets[EE] = acc;
    int t = 0;
    for (int e = 0; e < EE; ++e)
      for (int s = 0; s < counts[e]; s += 16) { tile_e[t] = e; tile_s[t] = offsets[e] + s; ++t; }
    *n_tiles = t;
  }
}

__global__ void k_scatter(const int* __restrict__ route_idx, const float* __restrict__ route_w,
                          int* cursor, int* __restrict__ perm_token,
                          float* __restrict__ perm_w, int* __restrict__ slot_of) {
  int g = blockIdx.x * blockDim.x + threadIdx.x;
  if (g >= NTOK) return;
#pragma unroll
  for (int k = 0; k < 2; ++k) {
    int e = route_idx[2 * g + k];
    int pos = atomicAdd(&cursor[e], 1);
    perm_token[pos] = g;
    perm_w[pos] = route_w[2 * g + k];
    slot_of[2 * g + k] = pos;
  }
}

// ------------- grouped expert MLP: out = (relu(A@W1+b1)@W2+b2)*w -----------
__global__ void __launch_bounds__(128)
k_expert(const __bf16* __restrict__ tok_bf, const __bf16* __restrict__ W1T,
         const float* __restrict__ b1, const __bf16* __restrict__ W2T,
         const float* __restrict__ b2, const int* __restrict__ tile_e,
         const int* __restrict__ tile_s, const int* __restrict__ n_tiles,
         const int* __restrict__ offsets, const int* __restrict__ perm_token,
         const float* __restrict__ perm_w, float* __restrict__ expert_out) {
  __shared__ __bf16 As[16 * HH];     // gathered token tile
  __shared__ __bf16 Hs[16 * FF];     // hidden tile (relu'd, bf16)
  __shared__ float  Ws[16];
  int wg = blockIdx.x;
  if (wg >= *n_tiles) return;
  int e = tile_e[wg], s0 = tile_s[wg];
  int valid = offsets[e + 1] - s0; if (valid > 16) valid = 16;
  int tid = threadIdx.x, wave = tid >> 5, lane = tid & 31;
  const __bf16* W1e = W1T + (size_t)e * FF * HH;
  const __bf16* W2e = W2T + (size_t)e * HH * FF;
  {   // prefetch this expert's weights (64KB each) into caches
    const char* w1p = (const char*)W1e;
    const char* w2p = (const char*)W2e;
#pragma unroll
    for (int q = 0; q < 4; ++q) {
      __builtin_prefetch(w1p + (size_t)tid * 512 + q * 128, 0, 3);
      __builtin_prefetch(w2p + (size_t)tid * 512 + q * 128, 0, 3);
    }
  }
  {   // gather A rows (pad with row 0's token, stores masked later)
    int row = tid >> 3, seg = tid & 7;
    int token = perm_token[(row < valid) ? (s0 + row) : s0];
    const __bf16* src = tok_bf + (size_t)token * HH + seg * 16;
    __bf16* dst = As + row * HH + seg * 16;
#if __has_builtin(__builtin_amdgcn_global_load_async_to_lds_b128)
    typedef __attribute__((address_space(1))) v4i* gv4p;
    typedef __attribute__((address_space(3))) v4i* lv4p;
    __builtin_amdgcn_global_load_async_to_lds_b128((gv4p)src, (lv4p)dst, 0, 0);
    __builtin_amdgcn_global_load_async_to_lds_b128((gv4p)src, (lv4p)dst, 16, 0);
#if __has_builtin(__builtin_amdgcn_s_wait_asynccnt)
    __builtin_amdgcn_s_wait_asynccnt(0);
#else
    asm volatile("s_wait_asynccnt 0x0" ::: "memory");
#endif
#else
#pragma unroll
    for (int j = 0; j < 16; ++j) dst[j] = src[j];
#endif
  }
  if (tid < 16) Ws[tid] = (tid < valid) ? perm_w[s0 + tid] : 0.f;
  __syncthreads();
  int col0 = lane & 15, rbase = (lane >> 4) << 3;
  // GEMM1: [16,128] @ [128,256]
  v8f acc[4];
#pragma unroll
  for (int j = 0; j < 4; ++j)
#pragma unroll
    for (int r = 0; r < 8; ++r) acc[j][r] = 0.f;
#pragma unroll
  for (int ks = 0; ks < 4; ++ks) {
    v16bf a = load_frag_a(As + ks * 32, HH);
#pragma unroll
    for (int j = 0; j < 4; ++j) {
      v16bf b = load_frag_bt(W1e + (wave * 4 + j) * 16 * HH + ks * 32, HH);
      acc[j] = wmma_bf16(a, b, acc[j]);
    }
  }
#pragma unroll
  for (int j = 0; j < 4; ++j)
#pragma unroll
    for (int r = 0; r < 8; ++r) {
      int row = rbase + r, col = (wave * 4 + j) * 16 + col0;
      float v = acc[j][r] + b1[e * FF + col];
      Hs[row * FF + col] = f2bf(v > 0.f ? v : 0.f);
    }
  __syncthreads();
  // GEMM2: [16,256] @ [256,128]
  v8f oacc[2];
#pragma unroll
  for (int j = 0; j < 2; ++j)
#pragma unroll
    for (int r = 0; r < 8; ++r) oacc[j][r] = 0.f;
#pragma unroll
  for (int ks = 0; ks < 8; ++ks) {
    v16bf a = load_frag_a(Hs + ks * 32, FF);
#pragma unroll
    for (int j = 0; j < 2; ++j) {
      v16bf b = load_frag_bt(W2e + (wave * 2 + j) * 16 * FF + ks * 32, FF);
      oacc[j] = wmma_bf16(a, b, oacc[j]);
    }
  }
#pragma unroll
  for (int j = 0; j < 2; ++j)
#pragma unroll
    for (int r = 0; r < 8; ++r) {
      int row = rbase + r;
      if (row < valid) {
        int col = (wave * 2 + j) * 16 + col0;
        float v = (oacc[j][r] + b2[e * HH + col]) * Ws[row];
        expert_out[(size_t)(s0 + row) * HH + col] = v;
      }
    }
}

// ------------- mix + residual + layernorm ----------------------------------
__global__ void __launch_bounds__(128)
k_final(const float* __restrict__ tok_f, const float* __restrict__ expert_out,
        const int* __restrict__ slot_of, const float* __restrict__ gamma,
        const float* __restrict__ beta, float* __restrict__ tokens_out) {
  int tid = threadIdx.x, wave = tid >> 5, lane = tid & 31;
  int g = blockIdx.x * 4 + wave;                      // one wave per token
  int s0 = slot_of[2 * g], s1 = slot_of[2 * g + 1];
  float x[4], sum = 0.f;
#pragma unroll
  for (int i = 0; i < 4; ++i) {
    int c = lane + i * 32;
    float v = tok_f[(size_t)g * HH + c]
            + expert_out[(size_t)s0 * HH + c]
            + expert_out[(size_t)s1 * HH + c];
    x[i] = v; sum += v;
  }
#pragma unroll
  for (int off = 16; off >= 1; off >>= 1) sum += __shfl_xor(sum, off, 32);
  float mu = sum * (1.f / HH);
  float vs = 0.f;
#pragma unroll
  for (int i = 0; i < 4; ++i) { float d = x[i] - mu; vs += d * d; }
#pragma unroll
  for (int off = 16; off >= 1; off >>= 1) vs += __shfl_xor(vs, off, 32);
  float inv = rsqrtf(vs * (1.f / HH) + 1e-5f);
#pragma unroll
  for (int i = 0; i < 4; ++i) {
    int c = lane + i * 32;
    tokens_out[(size_t)g * HH + c] = (x[i] - mu) * inv * gamma[c] + beta[c];
  }
}

// ------------- pooled mean (deterministic fixed-order reduction) -----------
__global__ void k_pool(const float* __restrict__ tokens_out, float* __restrict__ pooled) {
  int idx = blockIdx.x * blockDim.x + threadIdx.x;    // [0, B*H)
  if (idx >= NB * HH) return;
  int b = idx / HH, c = idx % HH;
  const float* p = tokens_out + (size_t)b * SEQ * HH + c;
  float s = 0.f;
  for (int i = 0; i < SEQ; ++i) s += p[(size_t)i * HH];
  pooled[idx] = s * (1.f / SEQ);
}

extern "C" void kernel_launch(void* const* d_in, const int* in_sizes, int n_in,
                              void* d_out, int out_size, void* d_ws, size_t ws_size,
                              hipStream_t stream) {
  const float* tokens_in = (const float*)d_in[0];
  const float* Wp   = (const float*)d_in[1];
  const float* bp   = (const float*)d_in[2];
  const float* Wq   = (const float*)d_in[3];
  const float* keys = (const float*)d_in[4];
  const float* W1   = (const float*)d_in[5];
  const float* b1   = (const float*)d_in[6];
  const float* W2   = (const float*)d_in[7];
  const float* b2   = (const float*)d_in[8];
  const float* gam  = (const float*)d_in[9];
  const float* bet  = (const float*)d_in[10];

  float* pooled     = (float*)d_out;                  // [B,H]
  float* tokens_out = (float*)d_out + NB * HH;        // [B,S,H]

  char* p = (char*)d_ws;
  auto alloc = [&](size_t bytes) { void* r = (void*)p; p += (bytes + 255) & ~(size_t)255; return r; };
  float*  tok_f      = (float*)alloc((size_t)NTOK * HH * 4);
  __bf16* tok_bf     = (__bf16*)alloc((size_t)NTOK * HH * 2);
  float*  expert_out = (float*)alloc((size_t)2 * NTOK * HH * 4);
  __bf16* WpB  = (__bf16*)alloc(HH * DD * 2);
  __bf16* WqB  = (__bf16*)alloc(HH * HH * 2);
  __bf16* keysB= (__bf16*)alloc(EE * HH * 2);
  __bf16* W1T  = (__bf16*)alloc((size_t)EE * FF * HH * 2);
  __bf16* W2T  = (__bf16*)alloc((size_t)EE * HH * FF * 2);
  int*    route_idx = (int*)alloc((size_t)NTOK * 2 * 4);
  float*  route_w   = (float*)alloc((size_t)NTOK * 2 * 4);
  int*    slot_of   = (int*)alloc((size_t)NTOK * 2 * 4);
  int*    perm_token= (int*)alloc((size_t)2 * NTOK * 4);
  float*  perm_w    = (float*)alloc((size_t)2 * NTOK * 4);
  int*    counts  = (int*)alloc(EE * 4);
  int*    offsets = (int*)alloc((EE + 1) * 4);
  int*    cursor  = (int*)alloc(EE * 4);
  int*    tile_e  = (int*)alloc(MAX_TILES * 4);
  int*    tile_s  = (int*)alloc(MAX_TILES * 4);
  int*    n_tiles = (int*)alloc(4);

  k_conv_small<<<(HH * DD + 255) / 256, 256, 0, stream>>>(Wp, Wq, keys, WpB, WqB, keysB, counts);
  k_t_w1<<<(EE * FF * HH + 255) / 256, 256, 0, stream>>>(W1, W1T);
  k_t_w2<<<(EE * HH * FF + 255) / 256, 256, 0, stream>>>(W2, W2T);
  k_proj<<<NTOK / 64, 128, 0, stream>>>(tokens_in, WpB, bp, tok_f, tok_bf);
  k_route<<<NTOK / 64, 128, 0, stream>>>(tok_bf, WqB, keysB, route_idx, route_w, counts);
  k_scan<<<1, 32, 0, stream>>>(counts, offsets, cursor, tile_e, tile_s, n_tiles);
  k_scatter<<<(NTOK + 255) / 256, 256, 0, stream>>>(route_idx, route_w, cursor, perm_token, perm_w, slot_of);
  k_expert<<<MAX_TILES, 128, 0, stream>>>(tok_bf, W1T, b1, W2T, b2, tile_e, tile_s, n_tiles,
                                          offsets, perm_token, perm_w, expert_out);
  k_final<<<NTOK / 4, 128, 0, stream>>>(tok_f, expert_out, slot_of, gam, bet, tokens_out);
  k_pool<<<(NB * HH + 255) / 256, 256, 0, stream>>>(tokens_out, pooled);
}
